// KIVILinear_76020921140199
// MI455X (gfx1250) — compile-verified, matrix operand order checked by reading, not run on hardware
//
#include <hip/hip_runtime.h>
#include <hip/hip_bf16.h>

typedef _Float16 v2h  __attribute__((ext_vector_type(2)));
typedef _Float16 v8h  __attribute__((ext_vector_type(8)));
typedef _Float16 v16h __attribute__((ext_vector_type(16)));
typedef float    v8f  __attribute__((ext_vector_type(8)));

// builtin argument types (GCC vector style, per hipcc diagnostics)
typedef int      g4i  __attribute__((vector_size(16)));
typedef unsigned u32x4 __attribute__((vector_size(16)));
typedef int      i32x8 __attribute__((vector_size(32)));
typedef int      i32x4 __attribute__((vector_size(16)));
typedef g4i __attribute__((address_space(1)))* g4i_gptr;
typedef g4i __attribute__((address_space(3)))* g4i_lptr;

#define K_DIM 4096      // IN_FEATURES (fixed by the reference)
#define GS    128       // quant group size
#define BM    64        // block tile M
#define BN    128       // block tile N
#define BK    32        // K per WMMA step
#define LDSK  40        // padded K stride in LDS (80B rows: 16B aligned, conflict-free)

#if __has_builtin(__builtin_amdgcn_tensor_load_to_lds) && \
    __has_builtin(__builtin_amdgcn_s_wait_tensorcnt)
#define USE_TDM 1
#else
#define USE_TDM 0
#endif

#if __has_builtin(__builtin_amdgcn_global_load_async_to_lds_b128) && \
    __has_builtin(__builtin_amdgcn_s_wait_asynccnt)
#define USE_ASYNC_LDS 1
#else
#define USE_ASYNC_LDS 0
#endif

// dequant 8 int4 nibbles -> v8h:  w[j] = s*q[j] - (1024*s + z)
// via fp16 magic bias: bitcast(0x6400 | q) == 1024 + q  (exact for 0<=q<16)
__device__ __forceinline__ v8h dq8(int q, _Float16 s, _Float16 zb) {
    union { v8h v; unsigned u[4]; } r;
    const v2h s2  = { s, s };
    const v2h zb2 = { zb, zb };
    #pragma unroll
    for (int d = 0; d < 4; ++d) {
        const int b = q >> (8 * d);
        const unsigned t = ((unsigned)(b & 0xF) | (((unsigned)(b & 0xF0)) << 12)) | 0x64006400u;
        const v2h hv = __builtin_bit_cast(v2h, t);
        const v2h w  = hv * s2 - zb2;               // one v_pk_fma_f16 per 2 nibbles
        r.u[d] = __builtin_bit_cast(unsigned, w);
    }
    return r.v;
}

__global__ __launch_bounds__(256) void kivi_w4a16_wmma(
    const _Float16* __restrict__ x,       // [M, K] fp16
    const int*      __restrict__ qweight, // [K/8, N] int32, 8 nibbles along K
    const float*    __restrict__ scales,  // [K/GS, N]
    const float*    __restrict__ zeros,   // [K/GS, N] (already scale*zero)
    _Float16*       __restrict__ out,     // [M, N] fp16
    int M, int N)
{
    __shared__ __align__(16) _Float16 As[2][BM][LDSK];  // 10 KB
    __shared__ __align__(16) _Float16 Bs[2][BN][LDSK];  // 20 KB

    const int tid  = threadIdx.x;
    const int lane = tid & 31;
    const int wave = tid >> 5;
    const int m0   = blockIdx.y * BM;
    const int n0   = blockIdx.x * BN;

    // dequant assignment: 4 qweight rows (32 K) x 128 cols, 2 int32 per thread
    const int qr = tid >> 6;            // 0..3
    const int nn = (tid & 63) * 2;      // local column pair

    // wave tile within block: 2 (M) x 4 (N) waves, each 32x32
    const int wm = (wave & 1) << 5;
    const int wn = (wave >> 1) << 5;
    const int hl = lane >> 4;           // K-half selector (WMMA 16-bit layout)
    const int ll = lane & 15;           // row (A) / col (B,D) within fragment

    v8f acc00 = {}, acc01 = {}, acc10 = {}, acc11 = {};

    _Float16 s0h = (_Float16)0.f, s1h = (_Float16)0.f;
    _Float16 zb0 = (_Float16)0.f, zb1 = (_Float16)0.f;  // 1024*s + z
    int gcur = -1;

    const int NSTEP = K_DIM / BK; // 128

    auto stage = [&](int step) {
        const int k0  = step * BK;
        const int buf = step & 1;
        // ---- A: 64x32 fp16 tile -> LDS (padded 40-fp16 rows) ----
#if USE_TDM
        if (wave == 0) {
            // Tensor DMA: one descriptor moves the whole tile, TDM applies the
            // 16-DWORD-row + 4-DWORD-pad layout (=> LDSK stride) in flight.
            const unsigned long long ga =
                (unsigned long long)(uintptr_t)(const void*)(x + (size_t)m0 * K_DIM + k0);
            const unsigned lds_off = (unsigned)(uintptr_t)(void*)&As[buf][0][0];
            u32x4 g0 = { 1u,                                    // count=1 (valid D#)
                         lds_off,                               // lds_addr (bytes)
                         (unsigned)(ga & 0xFFFFFFFFu),          // global_addr[31:0]
                         (unsigned)((ga >> 32) & 0x01FFFFFFu) | 0x80000000u }; // [56:32] | type=2
            i32x8 g1 = { (int)0x06D10000,      // data_size=2B, pad_en, interval=16dw, pad=4dw
                         (int)(4096u << 16),   // tensor_dim0 = 4096 (lo16 in [31:16])
                         (int)(64u << 16),     // tensor_dim0 hi | tensor_dim1=64
                         (int)(32u << 16),     // tensor_dim1 hi | tile_dim0=32
                         64,                   // tile_dim1=64 | tile_dim2=0
                         4096,                 // tensor_dim0_stride = K
                         0, 0 };
            i32x4 gz4 = { 0, 0, 0, 0 };
            i32x8 gz8 = { 0, 0, 0, 0, 0, 0, 0, 0 };
            __builtin_amdgcn_tensor_load_to_lds(g0, g1, gz4, gz4, gz8, 0);
        }
#else
        {
            const int row = tid >> 2, kc = tid & 3;
            const _Float16* gp = x + (size_t)(m0 + row) * K_DIM + k0 + kc * 8;
            _Float16* lp = &As[buf][row][kc * 8];
#if USE_ASYNC_LDS
            __builtin_amdgcn_global_load_async_to_lds_b128(
                (g4i_gptr)(uintptr_t)(const void*)gp,
                (g4i_lptr)(unsigned)(uintptr_t)(void*)lp,
                0, 0);
#else
            *(v8h*)lp = *(const v8h*)gp;
#endif
        }
#endif
        // ---- scales / zeros: reload once per 128-K group ----
        const int g = k0 >> 7;
        if (g != gcur) {
            gcur = g;
            const size_t sb = (size_t)g * N + n0 + nn;
            s0h = (_Float16)scales[sb];
            s1h = (_Float16)scales[sb + 1];
            zb0 = s0h * (_Float16)1024.0f + (_Float16)zeros[sb];
            zb1 = s1h * (_Float16)1024.0f + (_Float16)zeros[sb + 1];
        }
        // ---- B: dequant 2 int32 (16 nibbles) -> fp16, store transposed [n][k] ----
        {
            const size_t qb = (size_t)(k0 / 8 + qr) * N + n0 + nn;
            const int q0 = qweight[qb];
            const int q1 = qweight[qb + 1];
            *(v8h*)&Bs[buf][nn    ][qr * 8] = dq8(q0, s0h, zb0);
            *(v8h*)&Bs[buf][nn + 1][qr * 8] = dq8(q1, s1h, zb1);
        }
    };

    stage(0);

    for (int step = 0; step < NSTEP; ++step) {
#if USE_TDM
        if (wave == 0) __builtin_amdgcn_s_wait_tensorcnt(0);  // tile in LDS before signal
#elif USE_ASYNC_LDS
        __builtin_amdgcn_s_wait_asynccnt(0);
#endif
        __syncthreads();
        if (step + 1 < NSTEP) {
            if (step + 2 < NSTEP) {
                const int kp = (step + 2) * BK;
                __builtin_prefetch(qweight + (size_t)(kp / 8 + qr) * N + n0 + nn, 0, 3);
            }
            stage(step + 1);
        }
        const int buf = step & 1;

        // WMMA 16-bit fragment: elems 0..7 = K(hl*8 + 0..7), 8..15 = K(16 + hl*8 + 0..7)
        #define LOADF(arr, rowbase) ({                                              \
            v8h lo_ = *(const v8h*)&arr[buf][(rowbase)][hl * 8];                    \
            v8h hi_ = *(const v8h*)&arr[buf][(rowbase)][16 + hl * 8];               \
            __builtin_shufflevector(lo_, hi_, 0,1,2,3,4,5,6,7,                      \
                                              8,9,10,11,12,13,14,15); })

        v16h a0 = LOADF(As, wm + ll);
        v16h a1 = LOADF(As, wm + 16 + ll);
        v16h b0 = LOADF(Bs, wn + ll);
        v16h b1 = LOADF(Bs, wn + 16 + ll);
        #undef LOADF

        acc00 = __builtin_amdgcn_wmma_f32_16x16x32_f16(false, a0, false, b0,
                                                       (short)0, acc00, false, false);
        acc01 = __builtin_amdgcn_wmma_f32_16x16x32_f16(false, a0, false, b1,
                                                       (short)0, acc01, false, false);
        acc10 = __builtin_amdgcn_wmma_f32_16x16x32_f16(false, a1, false, b0,
                                                       (short)0, acc10, false, false);
        acc11 = __builtin_amdgcn_wmma_f32_16x16x32_f16(false, a1, false, b1,
                                                       (short)0, acc11, false, false);
    }

    // ---- epilogue: C/D layout -> fp16 global stores (coalesced across lanes) ----
    auto store_tile = [&](const v8f& c, int mt, int nt) {
        const int col  = n0 + wn + nt + ll;
        const int rowb = m0 + wm + mt + hl * 8;
        #pragma unroll
        for (int r = 0; r < 8; ++r)
            out[(size_t)(rowb + r) * N + col] = (_Float16)c[r];
    };
    store_tile(acc00, 0, 0);
    store_tile(acc01, 0, 16);
    store_tile(acc10, 16, 0);
    store_tile(acc11, 16, 16);
}

extern "C" void kernel_launch(void* const* d_in, const int* in_sizes, int n_in,
                              void* d_out, int out_size, void* d_ws, size_t ws_size,
                              hipStream_t stream) {
    const _Float16* x  = (const _Float16*)d_in[0];
    const int*      qw = (const int*)d_in[1];
    const float*    sc = (const float*)d_in[2];
    const float*    zr = (const float*)d_in[3];
    _Float16*       out = (_Float16*)d_out;

    const int M = in_sizes[0] / K_DIM;             // 256
    const int N = in_sizes[1] / (K_DIM / 8);       // 14336

    dim3 grid((unsigned)(N / BN), (unsigned)(M / BM));
    kivi_w4a16_wmma<<<grid, 256, 0, stream>>>(x, qw, sc, zr, out, M, N);
}
